// MultiHeadedAttention_9509057593556
// MI455X (gfx1250) — matrix operand order repaired and checked by reading, not compiled, not run on hardware
//
#include <hip/hip_runtime.h>
#include <hip/hip_bf16.h>

// Problem constants (match reference)
#define BB   2
#define SS   2048
#define DDIM 512
#define HH   8
#define DK   64

typedef __attribute__((ext_vector_type(16))) _Float16 v16h;
typedef __attribute__((ext_vector_type(8)))  _Float16 v8h;
typedef __attribute__((ext_vector_type(8)))  float    v8f;

__device__ __forceinline__ v8f wmma_f16(v16h a, v16h b, v8f c) {
  // D = A(16x32 f16) * B(32x16 f16) + C(16x16 f32)
  return __builtin_amdgcn_wmma_f32_16x16x32_f16(false, a, false, b, (short)0, c,
                                                false, false);
}

// A-matrix fragment: halves 0..15 = base[0..15]
// (lane = M row, laneHi selects K range {h*16..h*16+15})
__device__ __forceinline__ v16h load_a_frag(const _Float16* base) {
  v8h lo = *reinterpret_cast<const v8h*>(base);
  v8h hi = *reinterpret_cast<const v8h*>(base + 8);
  v16h a;
#pragma unroll
  for (int i = 0; i < 8; ++i) { a[i] = lo[i]; a[i + 8] = hi[i]; }
  return a;
}

// Async global->LDS copy of one 16B chunk (CDNA5 ASYNCcnt path).
// VDST = per-lane LDS byte offset, VADDR = 64-bit global address (GV mode).
__device__ __forceinline__ void async_copy_b128(unsigned lds_off,
                                                const void* gaddr) {
  asm volatile("global_load_async_to_lds_b128 %0, %1, off"
               :: "v"(lds_off), "v"(gaddr)
               : "memory");
}
__device__ __forceinline__ void wait_asynccnt_le4() {
  asm volatile("s_wait_asynccnt 0x4" ::: "memory");
}
__device__ __forceinline__ void wait_asynccnt_0() {
  asm volatile("s_wait_asynccnt 0x0" ::: "memory");
}

// ---------------------------------------------------------------- converts
__global__ void cvt_f32_to_f16(const float* __restrict__ in,
                               _Float16* __restrict__ out, int n) {
  for (int i = blockIdx.x * blockDim.x + threadIdx.x; i < n;
       i += gridDim.x * blockDim.x)
    out[i] = (_Float16)in[i];
}

// ---------------------------------------------------------------- WMMA GEMM
// C[M,N] = A[M,K] @ Bw[N,K]^T + bias[N]   (A,Bw f16 row-major, f32 accum)
// mode 0: store f16 to out16 as [b,h,s,dk]  (row=b*S+s, col=h*DK+dk)
// mode 2: store f16 to out16 as [b,h,dk,s]  (V transposed for PV B-operand)
// mode 3: store f32 to outf  as [row, col]  (final output)
__global__ __launch_bounds__(256) void gemm_wmma(
    const _Float16* __restrict__ A, const _Float16* __restrict__ Bw,
    const float* __restrict__ bias, _Float16* __restrict__ out16,
    float* __restrict__ outf, int M, int N, int K, int mode) {
  const int wave    = threadIdx.x >> 5;
  const int lane    = threadIdx.x & 31;
  const int lane_lo = lane & 15;
  const int laneHi  = lane >> 4;

  const int tile    = blockIdx.x * (blockDim.x >> 5) + wave;
  const int tiles_n = N >> 4;
  const int tm = tile / tiles_n;
  const int tn = tile - tm * tiles_n;

  const _Float16* arow = A  + (size_t)(tm * 16 + lane_lo) * K + laneHi * 16;
  const _Float16* brow = Bw + (size_t)(tn * 16 + lane_lo) * K + laneHi * 16;

  v8f c = {};
  for (int k0 = 0; k0 < K; k0 += 32) {
    v16h a = load_a_frag(arow + k0);
    v16h b = *reinterpret_cast<const v16h*>(brow + k0);
    c = wmma_f16(a, b, c);
  }

  const int   col = tn * 16 + lane_lo;
  const float bv  = bias[col];
  const int   h   = col / DK;
  const int   dkc = col - h * DK;

#pragma unroll
  for (int r = 0; r < 8; ++r) {
    const int   row = tm * 16 + laneHi * 8 + r;
    const float val = c[r] + bv;
    if (mode == 3) {
      outf[(size_t)row * DDIM + col] = val;
    } else {
      const int b = row / SS;
      const int s = row - b * SS;
      if (mode == 0)
        out16[(((size_t)b * HH + h) * SS + s) * DK + dkc] = (_Float16)val;
      else  // mode 2: transposed V
        out16[(((size_t)b * HH + h) * DK + dkc) * SS + s] = (_Float16)val;
    }
  }
}

// ---------------------------------------------------------------- attention
// One block = 4 waves sharing one (b,h); each wave owns 16 query rows.
// K/V tiles (32 keys) are staged into LDS once per block with
// GLOBAL_LOAD_ASYNC_TO_LDS_B128, double-buffered and overlapped with WMMA.
__global__ __launch_bounds__(128) void attn_wmma(
    const _Float16* __restrict__ Qh,   // [bh, s, dk]
    const _Float16* __restrict__ Kh,   // [bh, s, dk]
    const _Float16* __restrict__ Vt,   // [bh, dk, s]
    const float* __restrict__ gp,      // [b, q, k]
    const int* __restrict__ mask,      // [b, q, k]
    _Float16* __restrict__ Oh) {       // [b, s, D]
  // Padded rows: 72 halfs = 144B and 40 halfs = 80B keep 16B alignment and
  // spread the 16 per-lane rows across distinct bank groups.
  __shared__ __align__(16) _Float16 KtileS[2][32][72];
  __shared__ __align__(16) _Float16 VtileS[2][64][40];
  __shared__ __align__(16) _Float16 Plds[4][16][40];

  const int nqblk = SS / 64;
  const int qblk  = blockIdx.x % nqblk;
  const int bh    = blockIdx.x / nqblk;
  const int b     = bh / HH;
  const int h     = bh - b * HH;

  const int tid     = threadIdx.x;
  const int wave    = tid >> 5;
  const int lane    = tid & 31;
  const int lane_lo = lane & 15;
  const int laneHi  = lane >> 4;
  const int qbase   = qblk * 64 + wave * 16;

  // Issue the 4 async chunk-copies this thread owns for key step kb -> buf.
  auto stage = [&](int kb, int bufi) {
    // K tile: 32 key rows x 128B = 256 x 16B chunks, 2 per thread
    const char* kg = (const char*)(Kh + ((size_t)bh * SS + kb) * DK);
#pragma unroll
    for (int j = 0; j < 2; ++j) {
      const int e  = tid * 2 + j;
      const int kr = e >> 3, c = e & 7;
      async_copy_b128((unsigned)(size_t)&KtileS[bufi][kr][c * 8],
                      kg + kr * 128 + c * 16);
    }
    // V tile: 64 dk rows x 64B = 256 x 16B chunks, 2 per thread
    const char* vg = (const char*)(Vt + ((size_t)bh * DK) * SS + kb);
#pragma unroll
    for (int j = 0; j < 2; ++j) {
      const int e  = tid * 2 + j;
      const int vr = e >> 2, c = e & 3;
      async_copy_b128((unsigned)(size_t)&VtileS[bufi][vr][c * 8],
                      vg + (size_t)vr * SS * 2 + c * 16);
    }
  };

  // Q fragments (A layout), reused for all key tiles
  const _Float16* qptr =
      Qh + ((size_t)bh * SS + qbase + lane_lo) * DK + laneHi * 16;
  const v16h qf0 = load_a_frag(qptr);        // d = 0..31
  const v16h qf1 = load_a_frag(qptr + 32);   // d = 32..63

  v8f   oacc[4] = {};                        // 16q x 64dk f32 accumulator
  float mrow[8], lrow[8];
#pragma unroll
  for (int r = 0; r < 8; ++r) { mrow[r] = -3.0e38f; lrow[r] = 0.0f; }

  const float scale     = 0.125f;            // 1/sqrt(64)
  const int   qrow_lane = qbase + laneHi * 8;

  stage(0, 0);
  for (int kb = 0, it = 0; kb < SS; kb += 32, ++it) {
    const int cur = it & 1;
    if (kb + 32 < SS) {
      stage(kb + 32, cur ^ 1);   // fill other buffer behind current tile
      wait_asynccnt_le4();       // in-order: current tile's 4 asyncs landed
    } else {
      wait_asynccnt_0();
    }
    __syncthreads();             // tile visible to all 4 waves

    // ---- scores: two 16-key sub-tiles, each needs 2 WMMAs over d
    v8f c0 = {}, c1 = {};
    {
      v16h kf0 = *reinterpret_cast<const v16h*>(
          &KtileS[cur][lane_lo][laneHi * 16]);
      v16h kf1 = *reinterpret_cast<const v16h*>(
          &KtileS[cur][lane_lo][32 + laneHi * 16]);
      v16h kf2 = *reinterpret_cast<const v16h*>(
          &KtileS[cur][16 + lane_lo][laneHi * 16]);
      v16h kf3 = *reinterpret_cast<const v16h*>(
          &KtileS[cur][16 + lane_lo][32 + laneHi * 16]);
      c0 = wmma_f16(qf0, kf0, c0);
      c0 = wmma_f16(qf1, kf1, c0);
      c1 = wmma_f16(qf0, kf2, c1);
      c1 = wmma_f16(qf1, kf3, c1);
    }

    // ---- mask + scale, online softmax update
    float s0[8], s1[8];
#pragma unroll
    for (int r = 0; r < 8; ++r) {
      const int    q  = qrow_lane + r;
      const int    k0 = kb + lane_lo;
      const int    k1 = kb + 16 + lane_lo;
      const size_t mb = ((size_t)b * SS + q) * SS;
      const bool a0 = (mask[mb + k0] != 0) || (q == k0);
      const bool a1 = (mask[mb + k1] != 0) || (q == k1);
      s0[r] = a0 ? c0[r] * scale : -1.0e9f;
      s1[r] = a1 ? c1[r] * scale : -1.0e9f;
    }

#pragma unroll
    for (int r = 0; r < 8; ++r) {
      float t = fmaxf(s0[r], s1[r]);
      for (int off = 1; off < 16; off <<= 1)
        t = fmaxf(t, __shfl_xor(t, off, 32));     // row max (16-lane half)
      const float mnew  = fmaxf(mrow[r], t);
      const float alpha = __expf(mrow[r] - mnew);
      mrow[r] = mnew;

      const float p0 = __expf(s0[r] - mnew);
      const float p1 = __expf(s1[r] - mnew);
      float ps = p0 + p1;
      for (int off = 1; off < 16; off <<= 1)
        ps += __shfl_xor(ps, off, 32);            // row sum
      lrow[r] = lrow[r] * alpha + ps;

      // p_attn = softmax * group_prob (gp applied post-normalizer)
      const int    q  = qrow_lane + r;
      const size_t gb = ((size_t)b * SS + q) * SS;
      Plds[wave][laneHi * 8 + r][lane_lo]      =
          (_Float16)(p0 * gp[gb + kb + lane_lo]);
      Plds[wave][laneHi * 8 + r][16 + lane_lo] =
          (_Float16)(p1 * gp[gb + kb + 16 + lane_lo]);

#pragma unroll
      for (int nt = 0; nt < 4; ++nt) oacc[nt][r] *= alpha;
    }

    // ---- LDS transpose (C layout -> A layout), same-wave DS ordering
    const v16h pf = load_a_frag(&Plds[wave][lane_lo][laneHi * 16]);

    // ---- P @ V : 4 WMMAs over the 4 dk column tiles (B-operand from LDS)
#pragma unroll
    for (int nt = 0; nt < 4; ++nt) {
      v16h vf = *reinterpret_cast<const v16h*>(
          &VtileS[cur][nt * 16 + lane_lo][laneHi * 16]);
      oacc[nt] = wmma_f16(pf, vf, oacc[nt]);
    }

    __syncthreads();  // everyone done with buf `cur` before it is refilled
  }

  // ---- normalize and store O as f16 row-major [b, s, h*DK+dk]
#pragma unroll
  for (int r = 0; r < 8; ++r) {
    const float  inv = 1.0f / lrow[r];
    const int    q   = qrow_lane + r;
    const size_t ob  = ((size_t)b * SS + q) * DDIM + h * DK;
#pragma unroll
    for (int nt = 0; nt < 4; ++nt)
      Oh[ob + nt * 16 + lane_lo] = (_Float16)(oacc[nt][r] * inv);
  }
}

// ---------------------------------------------------------------- launcher
extern "C" void kernel_launch(void* const* d_in, const int* in_sizes, int n_in,
                              void* d_out, int out_size, void* d_ws,
                              size_t ws_size, hipStream_t stream) {
  const float* query = (const float*)d_in[0];
  const float* key   = (const float*)d_in[1];
  const float* value = (const float*)d_in[2];
  const float* gp    = (const float*)d_in[3];
  const int*   mask  = (const int*)d_in[4];
  const float* Wq    = (const float*)d_in[5];
  const float* bq    = (const float*)d_in[6];
  const float* Wk    = (const float*)d_in[7];
  const float* bk    = (const float*)d_in[8];
  const float* Wv    = (const float*)d_in[9];
  const float* bv    = (const float*)d_in[10];
  const float* Wo    = (const float*)d_in[11];
  const float* bo    = (const float*)d_in[12];

  const int nXD = BB * SS * DDIM;   // 2,097,152
  const int nW  = DDIM * DDIM;      //   262,144

  // Workspace layout (f16 buffers, ~30 MB total)
  char* ws = (char*)d_ws;
  size_t off = 0;
  auto take = [&](size_t bytes) {
    char* p = ws + off;
    off = (off + bytes + 255) & ~(size_t)255;
    return p;
  };
  _Float16* q16  = (_Float16*)take((size_t)nXD * 2);
  _Float16* k16  = (_Float16*)take((size_t)nXD * 2);
  _Float16* v16  = (_Float16*)take((size_t)nXD * 2);
  _Float16* wq16 = (_Float16*)take((size_t)nW * 2);
  _Float16* wk16 = (_Float16*)take((size_t)nW * 2);
  _Float16* wv16 = (_Float16*)take((size_t)nW * 2);
  _Float16* wo16 = (_Float16*)take((size_t)nW * 2);
  _Float16* Qh   = (_Float16*)take((size_t)nXD * 2);  // [bh,s,dk]
  _Float16* Kh   = (_Float16*)take((size_t)nXD * 2);  // [bh,s,dk]
  _Float16* Vt   = (_Float16*)take((size_t)nXD * 2);  // [bh,dk,s]
  _Float16* Oh   = (_Float16*)take((size_t)nXD * 2);  // [b,s,D]

  // 1) precision converts
  cvt_f32_to_f16<<<1024, 256, 0, stream>>>(query, q16, nXD);
  cvt_f32_to_f16<<<1024, 256, 0, stream>>>(key,   k16, nXD);
  cvt_f32_to_f16<<<1024, 256, 0, stream>>>(value, v16, nXD);
  cvt_f32_to_f16<<<256,  256, 0, stream>>>(Wq, wq16, nW);
  cvt_f32_to_f16<<<256,  256, 0, stream>>>(Wk, wk16, nW);
  cvt_f32_to_f16<<<256,  256, 0, stream>>>(Wv, wv16, nW);
  cvt_f32_to_f16<<<256,  256, 0, stream>>>(Wo, wo16, nW);

  // 2) projections: M=4096, N=512, K=512 -> 8192 16x16 tiles / 8 waves
  const int M = BB * SS, N = DDIM, K = DDIM;
  const int gemm_blocks = (M / 16) * (N / 16) / 8;  // 1024
  gemm_wmma<<<gemm_blocks, 256, 0, stream>>>(q16, wq16, bq, Qh, nullptr,
                                             M, N, K, 0);
  gemm_wmma<<<gemm_blocks, 256, 0, stream>>>(k16, wk16, bk, Kh, nullptr,
                                             M, N, K, 0);
  gemm_wmma<<<gemm_blocks, 256, 0, stream>>>(v16, wv16, bv, Vt, nullptr,
                                             M, N, K, 2);

  // 3) attention: B*H*(S/64) = 512 blocks x 4 waves
  attn_wmma<<<BB * HH * (SS / 64), 128, 0, stream>>>(Qh, Kh, Vt, gp, mask, Oh);

  // 4) output projection -> fp32 d_out
  gemm_wmma<<<gemm_blocks, 256, 0, stream>>>(Oh, wo16, bo, nullptr,
                                             (float*)d_out, M, N, K, 3);
}
